// EncoderStack_22308060136060
// MI455X (gfx1250) — matrix operand order repaired and checked by reading, not compiled, not run on hardware
//
#include <hip/hip_runtime.h>
#include <hip/hip_bf16.h>

// MI455X / gfx1250 encoder layer.
// GEMMs via v_wmma_f32_16x16x32_f16 (wave32, f32 acc), with double-buffered
// LDS tiles fed by GLOBAL_LOAD_ASYNC_TO_LDS_B128 (ASYNCcnt) when available.
// Compute-bound (~240 GFLOP); HBM traffic a few hundred MB at 23.3 TB/s.

typedef _Float16 half_t;
typedef __attribute__((ext_vector_type(8)))  _Float16 v8h;
typedef __attribute__((ext_vector_type(16))) _Float16 v16h;
typedef __attribute__((ext_vector_type(8)))  float    v8f;
typedef __attribute__((vector_size(16)))     int      vi4;   // b128 payload type

union V16H { v16h v; v8h h[2]; };

#if defined(__HIP_DEVICE_COMPILE__) && \
    __has_builtin(__builtin_amdgcn_global_load_async_to_lds_b128) && \
    __has_builtin(__builtin_amdgcn_s_wait_asynccnt)
#define USE_ASYNC_LDS 1
#else
#define USE_ASYNC_LDS 0
#endif

__device__ __forceinline__ void async_copy16(const half_t* g, half_t* l)
{
#if USE_ASYNC_LDS
    __builtin_amdgcn_global_load_async_to_lds_b128(
        (__attribute__((address_space(1))) vi4*)(void*)g,
        (__attribute__((address_space(3))) vi4*)(void*)l, 0, 0);
#else
    *(v8h*)l = *(const v8h*)g;
#endif
}

__device__ __forceinline__ void wait_lds_ready()
{
#if USE_ASYNC_LDS
    __builtin_amdgcn_s_wait_asynccnt(0);
#endif
}

// ---------------------------------------------------------------------------
// Generic WMMA GEMM:  C[m][n] = alpha * sum_k A[m][k] * Bt[n][k]  (+bias +res)
// A: MxK row-major f16 (lda), Bt: NxK row-major f16 (ldb) == B^T.
// Optional f32 and/or f16 outputs, bias (per n), residual (f32, ldc), ReLU.
// Batched over grid.z with (b,h) element strides: z -> (z/Hdiv, z%Hdiv).
// Double-buffered dynamic-LDS pipeline: tile s+1 streams into LDS (async
// DMA path on gfx1250) while tile s is consumed by WMMA.
// All dims are multiples of the tile sizes for this problem.
// ---------------------------------------------------------------------------
template<int BM, int BN, int WAVES_M, int WAVES_N, int WM, int WN>
__global__ __launch_bounds__(256) void gemm_wmma(
    const half_t* __restrict__ A, const half_t* __restrict__ Bt,
    float* __restrict__ Cf, half_t* __restrict__ Ch,
    const float* __restrict__ bias, const float* __restrict__ res,
    int M, int N, int K, int lda, int ldb, int ldc,
    float alpha, int relu,
    long sAb, long sAh, long sBb, long sBh, long sCb, long sCh, int Hdiv)
{
    static_assert(BM == WAVES_M * WM * 16, "BM");
    static_assert(BN == WAVES_N * WN * 16, "BN");
    static_assert(WAVES_M * WAVES_N == 8, "8 waves / 256 threads");
    constexpr int BK  = 64;        // two 16x16x32 k-steps per stage
    constexpr int PAD = 8;         // halves (16B) against LDS bank conflicts
    constexpr int BKP = BK + PAD;

    extern __shared__ __align__(16) char smem_raw[];
    half_t* sA = (half_t*)smem_raw;            // [2][BM][BKP]
    half_t* sB = sA + 2 * BM * BKP;            // [2][BN][BKP]

    const int z  = blockIdx.z;
    const int zb = z / Hdiv;
    const int zh = z % Hdiv;
    A  += (long)zb * sAb + (long)zh * sAh;
    Bt += (long)zb * sBb + (long)zh * sBh;
    const long coff = (long)zb * sCb + (long)zh * sCh;

    const int tid  = threadIdx.x;
    const int lane = tid & 31;
    const int wave = tid >> 5;
    const int wm   = wave % WAVES_M;
    const int wn   = wave / WAVES_M;
    const int lr   = lane & 15;
    const int lh   = lane >> 4;

    const int m0 = blockIdx.y * BM;
    const int n0 = blockIdx.x * BN;

    v8f acc[WM][WN] = {};

    constexpr int A_CHUNKS = BM * BK / 8 / 256;   // 16B chunks per thread
    constexpr int B_CHUNKS = BN * BK / 8 / 256;

    v8h rA[A_CHUNKS];   // register staging cache (fallback path only)
    v8h rB[B_CHUNKS];
    (void)rA; (void)rB;

    // Issue the global->LDS transfer for one tile stage.
    auto issue_load = [&](int buf, int k0) {
        #pragma unroll
        for (int i = 0; i < A_CHUNKS; i++) {
            int c = tid + i * 256;
            int r = c >> 3;            // BK/8 == 8 chunks per row
            int kc = (c & 7) * 8;
            const half_t* g = &A[(long)(m0 + r) * lda + k0 + kc];
#if USE_ASYNC_LDS
            async_copy16(g, &sA[((long)buf * BM + r) * BKP + kc]);
#else
            rA[i] = *(const v8h*)g;
#endif
        }
        #pragma unroll
        for (int i = 0; i < B_CHUNKS; i++) {
            int c = tid + i * 256;
            int r = c >> 3;
            int kc = (c & 7) * 8;
            const half_t* g = &Bt[(long)(n0 + r) * ldb + k0 + kc];
#if USE_ASYNC_LDS
            async_copy16(g, &sB[((long)buf * BN + r) * BKP + kc]);
#else
            rB[i] = *(const v8h*)g;
#endif
        }
    };
    // Commit register cache to LDS (no-op when async DMA does the write).
    auto commit_store = [&](int buf) {
#if !USE_ASYNC_LDS
        #pragma unroll
        for (int i = 0; i < A_CHUNKS; i++) {
            int c = tid + i * 256;
            int r = c >> 3;
            int kc = (c & 7) * 8;
            *(v8h*)&sA[((long)buf * BM + r) * BKP + kc] = rA[i];
        }
        #pragma unroll
        for (int i = 0; i < B_CHUNKS; i++) {
            int c = tid + i * 256;
            int r = c >> 3;
            int kc = (c & 7) * 8;
            *(v8h*)&sB[((long)buf * BN + r) * BKP + kc] = rB[i];
        }
#else
        (void)buf;
#endif
    };

    const int nstages = K / BK;

    issue_load(0, 0);
    commit_store(0);
    wait_lds_ready();
    __syncthreads();

    for (int s = 0; s < nstages; s++) {
        const int k0  = s * BK;
        const int buf = s & 1;

        if (s + 1 < nstages) issue_load(buf ^ 1, k0 + BK);

        // L2 prefetch of tile s+2 (global_prefetch_b8)
        if (k0 + 2 * BK < K) {
            __builtin_prefetch(&A[(long)(m0 + (tid & (BM - 1))) * lda + k0 + 2 * BK], 0, 2);
            __builtin_prefetch(&Bt[(long)(n0 + (tid & (BN - 1))) * ldb + k0 + 2 * BK], 0, 2);
        }

        const half_t* baseA = sA + (long)buf * BM * BKP;
        const half_t* baseB = sB + (long)buf * BN * BKP;

        #pragma unroll
        for (int kk = 0; kk < BK / 32; kk++) {
            V16H a[WM], b[WN];
            // A fragment (ISA 7.12.2, 16-bit A 16x32):
            //   lane m=lr, half lh: e<8 -> K = 8*lh+e ; e>=8 -> K = 16+8*lh+(e-8)
            #pragma unroll
            for (int tm = 0; tm < WM; tm++) {
                const half_t* p = baseA + (long)(wm * WM * 16 + tm * 16 + lr) * BKP + kk * 32;
                a[tm].h[0] = *(const v8h*)(p + 8 * lh);
                a[tm].h[1] = *(const v8h*)(p + 16 + 8 * lh);
            }
            // B fragment: lane col n=lr, e<8 -> K = 8*lh+e ; e>=8 -> K = 16+8*lh+(e-8) (Bs is N-major)
            #pragma unroll
            for (int tn = 0; tn < WN; tn++) {
                const half_t* p = baseB + (long)(wn * WN * 16 + tn * 16 + lr) * BKP + kk * 32;
                b[tn].h[0] = *(const v8h*)(p + 8 * lh);
                b[tn].h[1] = *(const v8h*)(p + 16 + 8 * lh);
            }
            #pragma unroll
            for (int tm = 0; tm < WM; tm++)
                #pragma unroll
                for (int tn = 0; tn < WN; tn++)
                    acc[tm][tn] = __builtin_amdgcn_wmma_f32_16x16x32_f16(
                        false, a[tm].v, false, b[tn].v,
                        (short)0, acc[tm][tn], false, false);
        }

        if (s + 1 < nstages) commit_store(buf ^ 1);
        wait_lds_ready();       // s_wait_asynccnt 0 (async path)
        __syncthreads();
    }

    // ---- epilogue: D layout reg r -> M = r + 8*lh, N = lr ----
    #pragma unroll
    for (int tm = 0; tm < WM; tm++) {
        #pragma unroll
        for (int tn = 0; tn < WN; tn++) {
            const int nn = n0 + wn * WN * 16 + tn * 16 + lr;
            const float bn = bias ? bias[nn] : 0.0f;
            #pragma unroll
            for (int r = 0; r < 8; r++) {
                const int mm = m0 + wm * WM * 16 + tm * 16 + 8 * lh + r;
                float v = alpha * acc[tm][tn][r] + bn;
                if (res)  v += res[(long)mm * ldc + nn];
                if (relu) v = fmaxf(v, 0.0f);
                const long off = coff + (long)mm * ldc + nn;
                if (Cf) Cf[off] = v;
                if (Ch) Ch[off] = (half_t)v;
            }
        }
    }
}

// ---------------------------------------------------------------------------
// Helpers
// ---------------------------------------------------------------------------
__global__ __launch_bounds__(256) void cast_f16_kernel(
    const float* __restrict__ in, half_t* __restrict__ out, long n)
{
    long i = (long)blockIdx.x * 1024 + threadIdx.x;
    #pragma unroll
    for (int j = 0; j < 4; j++) {
        long k = i + (long)j * 256;
        if (k < n) out[k] = (half_t)in[k];
    }
}

// W: (H=16, D, 64) fp32  ->  out: (N=H*64, D) f16, out[n][d] = W[h][d][k]
__global__ __launch_bounds__(256) void pack_head_kernel(
    const float* __restrict__ W, half_t* __restrict__ out, int D)
{
    int n = blockIdx.y;
    int d = blockIdx.x * 256 + threadIdx.x;
    int h = n >> 6, k = n & 63;
    out[(long)n * D + d] = (half_t)W[((long)h * D + d) * 64 + k];
}

// W: (R rows, C cols) fp32 row-major -> out: (C, R) f16 (transpose + cast)
__global__ __launch_bounds__(256) void pack_transpose_kernel(
    const float* __restrict__ W, half_t* __restrict__ out, int R, int C)
{
    int n = blockIdx.y;
    int k = blockIdx.x * 256 + threadIdx.x;
    out[(long)n * R + k] = (half_t)W[(long)k * C + n];
}

// vcat: (B*T, D) f16 concat-head layout -> vt: (B,H,64,T) f16
__global__ __launch_bounds__(256) void transpose_v_kernel(
    const half_t* __restrict__ vcat, half_t* __restrict__ vt, int T, int D)
{
    __shared__ half_t tile[64][65];
    int bh = blockIdx.y;
    int b = bh >> 4, h = bh & 15;
    int s0 = blockIdx.x * 64;
    #pragma unroll
    for (int i = 0; i < 16; i++) {
        int idx = threadIdx.x + i * 256;
        int sl = idx >> 6, vl = idx & 63;
        tile[sl][vl] = vcat[((long)(b * T + s0 + sl)) * D + h * 64 + vl];
    }
    __syncthreads();
    #pragma unroll
    for (int i = 0; i < 16; i++) {
        int idx = threadIdx.x + i * 256;
        int vl = idx >> 6, sl = idx & 63;
        vt[((long)bh * 64 + vl) * T + s0 + sl] = tile[sl][vl];
    }
}

// Column softmax (softmax over the QUERY axis q of scores[h][q][s]).
__global__ __launch_bounds__(256) void col_softmax_stats_kernel(
    const float* __restrict__ sc, float* __restrict__ cmax,
    float* __restrict__ csum, int T)
{
    int h = blockIdx.y;
    int s = blockIdx.x * 256 + threadIdx.x;
    const float* p = sc + (long)h * T * T + s;
    float m = -1e30f, sum = 0.0f;
    for (int q = 0; q < T; q++) {
        float v = p[(long)q * T];
        float nm = fmaxf(m, v);
        sum = sum * __expf(m - nm) + __expf(v - nm);
        m = nm;
    }
    cmax[h * T + s] = m;
    csum[h * T + s] = sum;
}

__global__ __launch_bounds__(256) void col_softmax_apply_kernel(
    const float* __restrict__ sc, const float* __restrict__ cmax,
    const float* __restrict__ csum, half_t* __restrict__ attn, int T)
{
    int h = blockIdx.y;
    int s = blockIdx.x * 256 + threadIdx.x;
    const float* p = sc + (long)h * T * T + s;
    half_t* o = attn + (long)h * T * T + s;
    float m = cmax[h * T + s];
    float rinv = 1.0f / csum[h * T + s];
    for (int q = 0; q < T; q++)
        o[(long)q * T] = (half_t)(__expf(p[(long)q * T] - m) * rinv);
}

// pseudo-norm: y - mean - std(ddof=1), per row of D=1024. In-place safe.
__global__ __launch_bounds__(256) void pseudo_norm_kernel(
    const float* __restrict__ in, float* __restrict__ outf,
    half_t* __restrict__ outh, int D)
{
    int row = blockIdx.x, tid = threadIdx.x;
    const float* p = in + (long)row * D;
    float v[4], s = 0.0f, s2 = 0.0f;
    #pragma unroll
    for (int i = 0; i < 4; i++) {
        v[i] = p[tid + i * 256];
        s += v[i];
        s2 += v[i] * v[i];
    }
    __shared__ float rs[256], rs2[256];
    rs[tid] = s; rs2[tid] = s2;
    __syncthreads();
    for (int w = 128; w > 0; w >>= 1) {
        if (tid < w) { rs[tid] += rs[tid + w]; rs2[tid] += rs2[tid + w]; }
        __syncthreads();
    }
    float mean = rs[0] / D;
    float var  = (rs2[0] - (float)D * mean * mean) / (float)(D - 1);
    float stdv = sqrtf(fmaxf(var, 0.0f));
    #pragma unroll
    for (int i = 0; i < 4; i++) {
        float o = v[i] - mean - stdv;
        long idx = (long)row * D + tid + i * 256;
        if (outf) outf[idx] = o;
        if (outh) outh[idx] = (half_t)o;
    }
}

// ---------------------------------------------------------------------------
extern "C" void kernel_launch(void* const* d_in, const int* in_sizes, int n_in,
                              void* d_out, int out_size, void* d_ws, size_t ws_size,
                              hipStream_t stream)
{
    (void)in_sizes; (void)n_in; (void)out_size; (void)ws_size;

    constexpr int Bz = 8, T = 1024, D = 1024, H = 16, DK = 64, FF = 4096;
    constexpr int M = Bz * T;  // 8192

    // Dynamic LDS for the two GEMM tile configs (double-buffered, PAD=8).
    constexpr size_t SH_A = (size_t)2 * (128 + 128) * 72 * sizeof(half_t); // 73728
    constexpr size_t SH_B = (size_t)2 * (128 + 64)  * 72 * sizeof(half_t); // 55296

    const float* x  = (const float*)d_in[0];
    const float* Wq = (const float*)d_in[1];
    const float* Wk = (const float*)d_in[2];
    const float* Wv = (const float*)d_in[3];
    const float* Wo = (const float*)d_in[4];
    const float* W1 = (const float*)d_in[5];
    const float* b1 = (const float*)d_in[6];
    const float* W2 = (const float*)d_in[7];
    const float* b2 = (const float*)d_in[8];
    float* out = (float*)d_out;

    // Workspace bump allocator (~248 MB total with aliasing).
    char* wsp = (char*)d_ws;
    size_t woff = 0;
    auto alloc = [&](size_t bytes) -> char* {
        char* p = wsp + woff;
        woff += (bytes + 255) & ~(size_t)255;
        return p;
    };
    half_t* xh   = (half_t*)alloc((size_t)M * D * 2);
    half_t* wqT  = (half_t*)alloc((size_t)D * D * 2);
    half_t* wkT  = (half_t*)alloc((size_t)D * D * 2);
    half_t* wvT  = (half_t*)alloc((size_t)D * D * 2);
    half_t* woT  = (half_t*)alloc((size_t)D * D * 2);
    half_t* w1h  = (half_t*)alloc((size_t)FF * D * 2);
    half_t* w2h  = (half_t*)alloc((size_t)D * FF * 2);
    half_t* qcat = (half_t*)alloc((size_t)M * D * 2);
    half_t* kcat = (half_t*)alloc((size_t)M * D * 2);
    half_t* vcat = (half_t*)alloc((size_t)M * D * 2);
    half_t* vt   = (half_t*)alloc((size_t)Bz * H * DK * T * 2);
    half_t* part = (half_t*)alloc((size_t)M * D * 2);
    float*  out1 = (float*) alloc((size_t)M * D * 4);
    float*  cmax = (float*) alloc((size_t)H * T * 4);
    float*  csum = (float*) alloc((size_t)H * T * 4);
    float*  scores = (float*)alloc((size_t)H * T * T * 4);   // 64 MB per-b chunk
    half_t* attn   = (half_t*)alloc((size_t)H * T * T * 2);  // 32 MB per-b chunk
    // Aliases (regions free by the time they're reused):
    half_t* hbuf = (half_t*)scores;  // FFN hidden (M*FF*2 == 64 MB)
    float*  out2 = (float*)attn;     // M*D*4 == 32 MB
    half_t* o1h  = part;             // normed out1 in f16 (part consumed first)

    // ---- weight/input packing to f16 ----
    cast_f16_kernel<<<(M * D) / 1024, 256, 0, stream>>>(x, xh, (long)M * D);
    pack_head_kernel<<<dim3(D / 256, D), 256, 0, stream>>>(Wq, wqT, D);
    pack_head_kernel<<<dim3(D / 256, D), 256, 0, stream>>>(Wk, wkT, D);
    pack_head_kernel<<<dim3(D / 256, D), 256, 0, stream>>>(Wv, wvT, D);
    pack_transpose_kernel<<<dim3(D / 256, D), 256, 0, stream>>>(Wo, woT, D, D);
    cast_f16_kernel<<<((long)FF * D) / 1024, 256, 0, stream>>>(W1, w1h, (long)FF * D);
    cast_f16_kernel<<<((long)D * FF) / 1024, 256, 0, stream>>>(W2, w2h, (long)D * FF);

    // ---- QKV projections: [M,D] x [D,D] -> f16 concat-head layout ----
    gemm_wmma<128,128,4,2,2,4><<<dim3(D/128, M/128, 1), 256, SH_A, stream>>>(
        xh, wqT, nullptr, qcat, nullptr, nullptr,
        M, D, D, D, D, D, 1.0f, 0, 0,0,0,0,0,0, 1);
    gemm_wmma<128,128,4,2,2,4><<<dim3(D/128, M/128, 1), 256, SH_A, stream>>>(
        xh, wkT, nullptr, kcat, nullptr, nullptr,
        M, D, D, D, D, D, 1.0f, 0, 0,0,0,0,0,0, 1);
    gemm_wmma<128,128,4,2,2,4><<<dim3(D/128, M/128, 1), 256, SH_A, stream>>>(
        xh, wvT, nullptr, vcat, nullptr, nullptr,
        M, D, D, D, D, D, 1.0f, 0, 0,0,0,0,0,0, 1);
    transpose_v_kernel<<<dim3(T / 64, Bz * H), 256, 0, stream>>>(vcat, vt, T, D);

    // ---- attention, chunked per batch index b (scores buffer reused) ----
    for (int b = 0; b < Bz; b++) {
        const half_t* Qb = qcat + (size_t)b * T * D;
        const half_t* Kb = kcat + (size_t)b * T * D;
        // scores[h][q][s] = (1/8) * Q[h,q,:64] . K[h,s,:64]
        gemm_wmma<128,128,4,2,2,4><<<dim3(T/128, T/128, H), 256, SH_A, stream>>>(
            Qb, Kb, scores, nullptr, nullptr, nullptr,
            T, T, DK, D, D, T, 0.125f, 0,
            0, 64, 0, 64, 0, (long)T * T, H);
        col_softmax_stats_kernel<<<dim3(T/256, H), 256, 0, stream>>>(scores, cmax, csum, T);
        col_softmax_apply_kernel<<<dim3(T/256, H), 256, 0, stream>>>(scores, cmax, csum, attn, T);
        // part[b,q,h*64+v] = attn[h,q,:] . Vt[h,v,:]
        gemm_wmma<128,64,8,1,1,4><<<dim3(1, T/128, H), 256, SH_B, stream>>>(
            attn, vt + (size_t)b * H * DK * T, nullptr, part + (size_t)b * T * D,
            nullptr, nullptr,
            T, DK, T, T, T, D, 1.0f, 0,
            0, (long)T * T, 0, (long)DK * T, 0, 64, H);
    }

    // ---- out1 = part @ Wo + x ; pseudo-norm (in place) + f16 copy ----
    gemm_wmma<128,128,4,2,2,4><<<dim3(D/128, M/128, 1), 256, SH_A, stream>>>(
        part, woT, out1, nullptr, nullptr, x,
        M, D, D, D, D, D, 1.0f, 0, 0,0,0,0,0,0, 1);
    pseudo_norm_kernel<<<M, 256, 0, stream>>>(out1, out1, o1h, D);

    // ---- FFN ----
    gemm_wmma<128,128,4,2,2,4><<<dim3(FF/128, M/128, 1), 256, SH_A, stream>>>(
        o1h, w1h, nullptr, hbuf, b1, nullptr,
        M, FF, D, D, D, FF, 1.0f, 1, 0,0,0,0,0,0, 1);
    gemm_wmma<128,128,4,2,2,4><<<dim3(D/128, M/128, 1), 256, SH_A, stream>>>(
        hbuf, w2h, out2, nullptr, b2, out1,
        M, D, FF, FF, FF, D, 1.0f, 0, 0,0,0,0,0,0, 1);
    pseudo_norm_kernel<<<M, 256, 0, stream>>>(out2, out, nullptr, D);
}
